// MarketNewsFusionWeightModel_67147518706141
// MI455X (gfx1250) — matrix operand order repaired and checked by reading, not compiled, not run on hardware
//
#include <hip/hip_runtime.h>
#include <hip/hip_bf16.h>
#include <stdint.h>

// ---------- CDNA5 WMMA types ----------
typedef __attribute__((ext_vector_type(16))) _Float16 v16h;
typedef __attribute__((ext_vector_type(8)))  _Float16 v8h;
typedef __attribute__((ext_vector_type(8)))  float    v8f;

#define WMMA(a, b, c) __builtin_amdgcn_wmma_f32_16x16x32_f16(false, (a), false, (b), (short)0, (c), false, false)

__device__ __forceinline__ float sigm(float x) { return 1.f / (1.f + __expf(-x)); }

// Model dims
#define BB 128
#define TT 256
#define DD 64
#define SS 100
#define TNN 128
#define DFFN 2048

// =====================================================================
// Generic batched GEMM:  Y[m,n] = act( sum_k X[m,k] * W[n,k] + bias[n] )
// X f16 (M x K, row stride ldx), W f16 (N x K, row stride ldw).
// Each wave32 computes one 16x16 tile via v_wmma_f32_16x16x32_f16.
// act: 0=none 1=relu 2=sigmoid. Y (f32) and/or Yh (f16) may be null.
// =====================================================================
__global__ __launch_bounds__(256) void k_gemm(
    const _Float16* __restrict__ X, const _Float16* __restrict__ W,
    const float* __restrict__ bias, float* __restrict__ Y, _Float16* __restrict__ Yh,
    int M, int N, int K,
    long long ldx, long long ldw, long long ldy,
    long long bsX, long long bsW, long long bsB, long long bsY, int act)
{
  int wave = threadIdx.x >> 5, lane = threadIdx.x & 31;
  int tilesN = N >> 4;
  int tile = blockIdx.x * 8 + wave;
  if (tile >= (M >> 4) * tilesN) return;
  int tm = tile / tilesN, tn = tile % tilesN;
  long long bz = blockIdx.z;
  const _Float16* Xp = X + bsX * bz;
  const _Float16* Wp = W + bsW * bz;
  int m0 = tm << 4, n0 = tn << 4;
  int ml = lane & 15, hi = lane >> 4;

  v8f acc = {};
  for (int kb = 0; kb < K; kb += 32) {
    // A: lane holds row m0+ml, K = kb + hi*8 + i (i<8) and kb+16+hi*8+(i-8)
    v16h a;
    const _Float16* ap = Xp + (long long)(m0 + ml) * ldx + kb + hi * 8;
    ((v8h*)&a)[0] = *(const v8h*)ap;
    ((v8h*)&a)[1] = *(const v8h*)(ap + 16);
    // B: lane holds col n0+ml, K = kb + hi*8 + i (i<8) and kb+16+hi*8+(i-8)
    v16h b;
    const _Float16* bp = Wp + (long long)(n0 + ml) * ldw + kb + hi * 8;
    ((v8h*)&b)[0] = *(const v8h*)bp;
    ((v8h*)&b)[1] = *(const v8h*)(bp + 16);
    acc = WMMA(a, b, acc);
  }
  float bv = bias ? bias[bsB * bz + n0 + ml] : 0.f;
  float* ap = (float*)&acc;
  for (int r = 0; r < 8; ++r) {
    float v = ap[r] + bv;
    if (act == 1) v = fmaxf(v, 0.f);
    else if (act == 2) v = sigm(v);
    long long off = (long long)(m0 + r + hi * 8) * ldy + bsY * bz + (n0 + ml);
    if (Y)  Y[off]  = v;
    if (Yh) Yh[off] = (_Float16)v;
  }
}

// =====================================================================
// Fused transformer attention: per (16-query tile, head, batch).
// Scores via WMMA (hd=16 zero-padded to K=32), softmax in LDS, P@V via WMMA.
// =====================================================================
__global__ __launch_bounds__(32) void k_attn(const _Float16* __restrict__ qkvh,
                                             _Float16* __restrict__ oh)
{
  __shared__ float sS[16 * 256];
  int lane = threadIdx.x;
  int m0 = blockIdx.x << 4;
  int h = blockIdx.y;
  int b = blockIdx.z;
  int ml = lane & 15, hi = lane >> 4;
  const _Float16* base = qkvh + (size_t)b * TT * 192;
  const _Float16* qp = base + h * 16;
  const _Float16* kp = base + 64 + h * 16;
  const _Float16* vp = base + 128 + h * 16;

  // A = Q tile 16x16 (K 16..31 = 0)
  v16h aq = {};
  ((v8h*)&aq)[0] = *(const v8h*)(qp + (size_t)(m0 + ml) * 192 + hi * 8);

  for (int nt = 0; nt < 16; ++nt) {
    v16h bk = {};
    {
      const _Float16* bp = kp + (size_t)(nt * 16 + ml) * 192;
      ((v8h*)&bk)[0] = *(const v8h*)(bp + hi * 8);
    }
    v8f c = {};
    c = WMMA(aq, bk, c);
    float* cp = (float*)&c;
    for (int r = 0; r < 8; ++r)
      sS[(r + hi * 8) * 256 + nt * 16 + ml] = cp[r] * 0.25f;  // 1/sqrt(16)
  }
  __syncthreads();
  // softmax over 256 keys; 2 lanes per query row
  {
    int row = lane >> 1, half = lane & 1;
    float* p = sS + row * 256 + half * 128;
    float mx = -3.4e38f;
    for (int j = 0; j < 128; ++j) mx = fmaxf(mx, p[j]);
    mx = fmaxf(mx, __shfl_xor(mx, 1));
    float sum = 0.f;
    for (int j = 0; j < 128; ++j) { float e = __expf(p[j] - mx); p[j] = e; sum += e; }
    sum += __shfl_xor(sum, 1);
    float inv = 1.f / sum;
    for (int j = 0; j < 128; ++j) p[j] *= inv;
  }
  __syncthreads();
  // O = P (16x256) @ V (256x16), K chunks of 32
  v8f o = {};
  for (int kc = 0; kc < 8; ++kc) {
    v16h pa;
    _Float16* pap = (_Float16*)&pa;
    const float* sp = sS + ml * 256 + kc * 32;
    for (int i = 0; i < 8; ++i) pap[i]     = (_Float16)sp[hi * 8 + i];
    for (int i = 0; i < 8; ++i) pap[8 + i] = (_Float16)sp[16 + hi * 8 + i];
    v16h vb;
    _Float16* vbp = (_Float16*)&vb;
    const _Float16* vv = vp + (size_t)(kc * 32 + hi * 8) * 192 + ml;
    for (int i = 0; i < 16; ++i) vbp[i] = vv[(size_t)(i < 8 ? i : i + 8) * 192];
    o = WMMA(pa, vb, o);
  }
  float* op = (float*)&o;
  for (int r = 0; r < 8; ++r)
    oh[((size_t)b * TT + m0 + r + hi * 8) * 64 + h * 16 + ml] = (_Float16)op[r];
}

// =====================================================================
// Fused FFN: y = relu(x@W1^T + b1)@W2^T + b2, hidden never hits HBM.
// One wave per 16-row tile; 16x32 hidden chunk staged in private LDS
// (same-wave DS ops are in-order on CDNA5) to re-shape into A-operand.
// =====================================================================
__global__ __launch_bounds__(256) void k_ffn(
    const _Float16* __restrict__ X, const _Float16* __restrict__ W1,
    const float* __restrict__ b1, const _Float16* __restrict__ W2,
    const float* __restrict__ b2, float* __restrict__ Y, int M)
{
  __shared__ float sH[8][16 * 32];
  int wave = threadIdx.x >> 5, lane = threadIdx.x & 31;
  int m0 = (blockIdx.x * 8 + wave) << 4;
  if (m0 >= M) return;
  int ml = lane & 15, hi = lane >> 4;
  float* sh = sH[wave];

  // X tile (16x64) -> two A operands
  v16h a0, a1;
  const _Float16* xp = X + (long long)(m0 + ml) * 64 + hi * 8;
  ((v8h*)&a0)[0] = *(const v8h*)xp;
  ((v8h*)&a0)[1] = *(const v8h*)(xp + 16);
  ((v8h*)&a1)[0] = *(const v8h*)(xp + 32);
  ((v8h*)&a1)[1] = *(const v8h*)(xp + 48);

  v8f y[4] = {};
  for (int c = 0; c < DFFN / 32; ++c) {
    // hidden chunk 16x32 = two 16x16 WMMA tiles
    for (int jt = 0; jt < 2; ++jt) {
      int j = c * 32 + jt * 16 + ml;
      const _Float16* wp = W1 + (long long)j * 64 + hi * 8;
      v16h bA, bB;
      ((v8h*)&bA)[0] = *(const v8h*)wp;
      ((v8h*)&bA)[1] = *(const v8h*)(wp + 16);
      ((v8h*)&bB)[0] = *(const v8h*)(wp + 32);
      ((v8h*)&bB)[1] = *(const v8h*)(wp + 48);
      v8f hc = {};
      hc = WMMA(a0, bA, hc);
      hc = WMMA(a1, bB, hc);
      float bb = b1[c * 32 + jt * 16 + ml];
      float* hp = (float*)&hc;
      for (int r = 0; r < 8; ++r)
        sh[(r + hi * 8) * 32 + jt * 16 + ml] = fmaxf(hp[r] + bb, 0.f);
    }
    // transpose-read hidden chunk as A operand (f32 -> f16)
    v16h pa;
    _Float16* pap = (_Float16*)&pa;
    const float* sp = sh + ml * 32;
    for (int i = 0; i < 8; ++i) pap[i]     = (_Float16)sp[hi * 8 + i];
    for (int i = 0; i < 8; ++i) pap[8 + i] = (_Float16)sp[16 + hi * 8 + i];
    // accumulate into 4 output n-tiles (N=64)
    for (int nt = 0; nt < 4; ++nt) {
      const _Float16* wp = W2 + (long long)(nt * 16 + ml) * DFFN + c * 32 + hi * 8;
      v16h wb;
      ((v8h*)&wb)[0] = *(const v8h*)wp;
      ((v8h*)&wb)[1] = *(const v8h*)(wp + 16);
      y[nt] = WMMA(pa, wb, y[nt]);
    }
  }
  for (int nt = 0; nt < 4; ++nt) {
    float bb = b2[nt * 16 + ml];
    float* yp = (float*)&y[nt];
    for (int r = 0; r < 8; ++r)
      Y[(long long)(m0 + r + hi * 8) * 64 + nt * 16 + ml] = yp[r] + bb;
  }
}

// =====================================================================
// Residual add + LayerNorm over D=64; one wave per row; writes f32+f16.
// =====================================================================
__global__ __launch_bounds__(256) void k_add_ln(
    const float* __restrict__ X, const float* __restrict__ R,
    const float* __restrict__ g, const float* __restrict__ bta,
    float* __restrict__ Y, _Float16* __restrict__ Yh, int rows)
{
  int wave = threadIdx.x >> 5, lane = threadIdx.x & 31;
  int row = blockIdx.x * 8 + wave;
  if (row >= rows) return;
  long long base = (long long)row * 64;
  float a0 = X[base + lane * 2]     + R[base + lane * 2];
  float a1 = X[base + lane * 2 + 1] + R[base + lane * 2 + 1];
  float s = a0 + a1, s2 = a0 * a0 + a1 * a1;
  for (int d = 1; d < 32; d <<= 1) { s += __shfl_xor(s, d); s2 += __shfl_xor(s2, d); }
  float mean = s * (1.f / 64.f);
  float var = s2 * (1.f / 64.f) - mean * mean;
  float inv = rsqrtf(var + 1e-5f);
  float y0 = (a0 - mean) * inv * g[lane * 2]     + bta[lane * 2];
  float y1 = (a1 - mean) * inv * g[lane * 2 + 1] + bta[lane * 2 + 1];
  Y[base + lane * 2] = y0;       Y[base + lane * 2 + 1] = y1;
  Yh[base + lane * 2] = (_Float16)y0; Yh[base + lane * 2 + 1] = (_Float16)y1;
}

// =====================================================================
// LSTM: one wave owns 16 batch rows for the full 128-step recurrence.
// h (f16) and c (f32) live in LDS; per step, 16 n-tiles of h@W_hh^T via
// WMMA, then gate math entirely in-lane (i/f/g/o land in acc[nt+4k]).
// gx = x_t@W_ih^T + b_ih was precomputed by k_gemm.
// =====================================================================
__global__ __launch_bounds__(32) void k_lstm(
    const float* __restrict__ gx, const _Float16* __restrict__ Whh,
    const float* __restrict__ bhh, float* __restrict__ emb,
    _Float16* __restrict__ embh)
{
  __shared__ float    cst[16 * 64];
  __shared__ _Float16 hh[16 * 64];
  int lane = threadIdx.x;
  int m0 = blockIdx.x * 16;
  int ml = lane & 15, hi = lane >> 4;
  for (int i = lane; i < 16 * 64; i += 32) { cst[i] = 0.f; hh[i] = (_Float16)0.f; }
  __syncthreads();

  for (int t = 0; t < TNN; ++t) {
    // A operands from LDS h (16x64)
    v16h a0, a1;
    const _Float16* hp = hh + ml * 64 + hi * 8;
    ((v8h*)&a0)[0] = *(const v8h*)hp;
    ((v8h*)&a0)[1] = *(const v8h*)(hp + 16);
    ((v8h*)&a1)[0] = *(const v8h*)(hp + 32);
    ((v8h*)&a1)[1] = *(const v8h*)(hp + 48);
    v8f acc[16];
    for (int nt = 0; nt < 16; ++nt) {
      const _Float16* wp = Whh + (long long)(nt * 16 + ml) * 64 + hi * 8;
      v16h bA, bB;
      ((v8h*)&bA)[0] = *(const v8h*)wp;
      ((v8h*)&bA)[1] = *(const v8h*)(wp + 16);
      ((v8h*)&bB)[0] = *(const v8h*)(wp + 32);
      ((v8h*)&bB)[1] = *(const v8h*)(wp + 48);
      v8f z = {};
      z = WMMA(a0, bA, z);
      z = WMMA(a1, bB, z);
      acc[nt] = z;
    }
    // gates: cell (m, j) uses acc[nt], acc[nt+4], acc[nt+8], acc[nt+12]
    for (int nt = 0; nt < 4; ++nt) {
      for (int r = 0; r < 8; ++r) {
        int mrel = r + hi * 8;
        long long grow = ((long long)(m0 + mrel) * TNN + t) * 256;
        int j = nt * 16 + ml;
        float gi = ((float*)&acc[nt])[r]      + gx[grow + j]       + bhh[j];
        float gf = ((float*)&acc[nt + 4])[r]  + gx[grow + j + 64]  + bhh[j + 64];
        float gg = ((float*)&acc[nt + 8])[r]  + gx[grow + j + 128] + bhh[j + 128];
        float go = ((float*)&acc[nt + 12])[r] + gx[grow + j + 192] + bhh[j + 192];
        int cell = mrel * 64 + j;
        float cc = sigm(gf) * cst[cell] + sigm(gi) * tanhf(gg);
        cst[cell] = cc;
        hh[cell] = (_Float16)(sigm(go) * tanhf(cc));
      }
    }
    __syncthreads();
  }
  for (int i = lane; i < 16 * 64; i += 32) {
    float v = (float)hh[i];
    emb[(long long)m0 * 64 + i] = v;
    embh[(long long)m0 * 64 + i] = (_Float16)v;
  }
}

// ---------------- small elementwise kernels ----------------
__global__ void k_cvt(const float* __restrict__ src, _Float16* __restrict__ dst,
                      int rows, int scols, int dcols)
{
  size_t idx = (size_t)blockIdx.x * 256 + threadIdx.x;
  if (idx >= (size_t)rows * dcols) return;
  int r = (int)(idx / dcols), c = (int)(idx % dcols);
  dst[idx] = (_Float16)(c < scols ? src[(size_t)r * scols + c] : 0.f);
}

__global__ void k_pack_news(const float* __restrict__ news, const float* __restrict__ cnt,
                            _Float16* __restrict__ dst)
{
  size_t idx = (size_t)blockIdx.x * 256 + threadIdx.x;
  if (idx >= (size_t)BB * TNN * 896) return;
  int r = (int)(idx / 896), c = (int)(idx % 896);
  float v = 0.f;
  if (c < 768)       v = news[(size_t)r * 768 + c];
  else if (c < 868)  v = cnt[(size_t)r * 100 + (c - 768)];
  dst[idx] = (_Float16)v;
}

__global__ void k_pe(float* __restrict__ x, _Float16* __restrict__ xh)
{
  size_t idx = (size_t)blockIdx.x * 256 + threadIdx.x;
  if (idx >= (size_t)BB * TT * 64) return;
  int t = (int)((idx >> 6) % TT);
  int d = (int)(idx & 63);
  float freq = __expf(-(float)(d & ~1) * 0.14391157f);  // ln(10000)/64
  float ang = (float)t * freq;
  float pe = (d & 1) ? __cosf(ang) : __sinf(ang);
  float v = x[idx] + pe;
  x[idx] = v;
  xh[idx] = (_Float16)v;
}

__global__ void k_extract(const float* __restrict__ x, float* __restrict__ e,
                          _Float16* __restrict__ eh)
{
  int idx = blockIdx.x * 256 + threadIdx.x;
  if (idx >= BB * 64) return;
  int b = idx / 64, d = idx % 64;
  float v = x[((size_t)b * TT + (TT - 1)) * 64 + d];
  e[idx] = v;
  eh[idx] = (_Float16)v;
}

__global__ void k_fuse_attn(const float* __restrict__ Q, const float* __restrict__ K,
                            const float* __restrict__ V, _Float16* __restrict__ atth)
{
  int idx = blockIdx.x * 256 + threadIdx.x;
  if (idx >= BB * SS * 4) return;
  int h = idx & 3, bs = idx >> 2;
  const float* q = Q + (size_t)bs * 64 + h * 16;
  const float* k = K + (size_t)bs * 64;
  const float* v = V + (size_t)bs * 64;
  float sc[4];
  for (int g = 0; g < 4; ++g) {
    float s = 0.f;
    for (int d = 0; d < 16; ++d) s += q[d] * k[g * 16 + d];
    sc[g] = s * 0.25f;  // 1/sqrt(16)
  }
  float mx = fmaxf(fmaxf(sc[0], sc[1]), fmaxf(sc[2], sc[3]));
  float e[4], sum = 0.f;
  for (int g = 0; g < 4; ++g) { e[g] = __expf(sc[g] - mx); sum += e[g]; }
  float inv = 1.f / sum;
  for (int d = 0; d < 16; ++d) {
    float o = 0.f;
    for (int g = 0; g < 4; ++g) o += e[g] * inv * v[g * 16 + d];
    atth[(size_t)bs * 64 + h * 16 + d] = (_Float16)o;
  }
}

__global__ void k_gate_mul(const float* __restrict__ u, const float* __restrict__ g,
                           _Float16* __restrict__ p)
{
  int idx = blockIdx.x * 256 + threadIdx.x;
  if (idx >= BB * SS * 64) return;
  p[idx] = (_Float16)(u[idx] * g[idx]);
}

__global__ void k_head(const float* __restrict__ fused, const float* __restrict__ Wh,
                       const float* __restrict__ bh, float* __restrict__ out)
{
  int idx = blockIdx.x * 256 + threadIdx.x;
  if (idx >= BB * SS) return;
  int s = idx % SS;
  const float* f = fused + (size_t)idx * 64;   // fused already ReLU'd by GEMM act=1
  float acc = bh[s];
  for (int o = 0; o < 64; ++o) acc += f[o] * Wh[s * 64 + o];
  out[idx] = acc;
}

// ---------------- host side ----------------
static void gemm(hipStream_t s, const _Float16* X, const _Float16* W, const float* bias,
                 float* Y, _Float16* Yh, int M, int N, int K,
                 long long ldx, long long ldw, long long ldy,
                 long long bsX, long long bsW, long long bsB, long long bsY,
                 int batch, int act)
{
  int tiles = (M / 16) * (N / 16);
  dim3 g((tiles + 7) / 8, 1, batch);
  k_gemm<<<g, 256, 0, s>>>(X, W, bias, Y, Yh, M, N, K, ldx, ldw, ldy, bsX, bsW, bsB, bsY, act);
}
static void cvt(hipStream_t s, const float* src, _Float16* dst, int rows, int sc, int dc)
{
  size_t n = (size_t)rows * dc;
  k_cvt<<<(int)((n + 255) / 256), 256, 0, s>>>(src, dst, rows, sc, dc);
}

extern "C" void kernel_launch(void* const* d_in, const int* in_sizes, int n_in,
                              void* d_out, int out_size, void* d_ws, size_t ws_size,
                              hipStream_t stream)
{
  (void)in_sizes; (void)n_in; (void)out_size; (void)ws_size;
  const float* ts_input  = (const float*)d_in[0];
  const float* count_in  = (const float*)d_in[1];
  const float* news_in   = (const float*)d_in[2];
  const float* W_inp = (const float*)d_in[3];  const float* b_inp = (const float*)d_in[4];
  const float* Wqkv  = (const float*)d_in[5];  const float* bqkv  = (const float*)d_in[6];
  const float* Wo    = (const float*)d_in[7];  const float* bo    = (const float*)d_in[8];
  const float* ln1_g = (const float*)d_in[9];  const float* ln1_b = (const float*)d_in[10];
  const float* W1    = (const float*)d_in[11]; const float* b1    = (const float*)d_in[12];
  const float* W2    = (const float*)d_in[13]; const float* b2    = (const float*)d_in[14];
  const float* ln2_g = (const float*)d_in[15]; const float* ln2_b = (const float*)d_in[16];
  const float* W_news = (const float*)d_in[17]; const float* b_news = (const float*)d_in[18];
  const float* W_ih  = (const float*)d_in[19]; const float* W_hh  = (const float*)d_in[20];
  const float* b_ih  = (const float*)d_in[21]; const float* b_hh  = (const float*)d_in[22];
  const float* Wq = (const float*)d_in[23]; const float* bq = (const float*)d_in[24];
  const float* Wk = (const float*)d_in[25]; const float* bk = (const float*)d_in[26];
  const float* Wv = (const float*)d_in[27]; const float* bv = (const float*)d_in[28];
  const float* Wg = (const float*)d_in[29]; const float* bg = (const float*)d_in[30];
  const float* Wu = (const float*)d_in[31]; const float* bu = (const float*)d_in[32];
  const float* Wf = (const float*)d_in[33]; const float* bf = (const float*)d_in[34];
  const float* W_head = (const float*)d_in[35]; const float* b_head = (const float*)d_in[36];
  float* out = (float*)d_out;

  // ---- carve workspace ----
  char* p = (char*)d_ws;
  auto cvH = [&](size_t n) { _Float16* r = (_Float16*)p; p += ((n * 2 + 255) & ~(size_t)255); return r; };
  auto cvF = [&](size_t n) { float* r = (float*)p; p += ((n * 4 + 255) & ~(size_t)255); return r; };

  const size_t MT = (size_t)BB * TT;       // 32768
  const size_t MN = (size_t)BB * TNN;      // 16384
  _Float16* tsh     = cvH(MT * 32);
  _Float16* wiph    = cvH(64 * 32);
  _Float16* xh      = cvH(MT * 64);
  _Float16* wqkvh   = cvH(2 * 192 * 64);
  _Float16* qkvh    = cvH(MT * 192);
  _Float16* ohb     = cvH(MT * 64);
  _Float16* woh     = cvH(2 * 64 * 64);
  _Float16* w1h     = cvH(2 * DFFN * 64);
  _Float16* w2h     = cvH(2 * 64 * DFFN);
  _Float16* newsh   = cvH(MN * 896);
  _Float16* wnewsh  = cvH(64 * 896);
  _Float16* nprojh  = cvH(MN * 64);
  _Float16* wihh    = cvH(256 * 64);
  _Float16* whhh    = cvH(256 * 64);
  _Float16* tsembh  = cvH(BB * 64);
  _Float16* nembh   = cvH(BB * 64);
  _Float16* wqh = cvH((size_t)SS * 64 * 64); _Float16* wkh = cvH((size_t)SS * 64 * 64);
  _Float16* wvh = cvH((size_t)SS * 64 * 64); _Float16* wgh = cvH((size_t)SS * 64 * 64);
  _Float16* wuh = cvH((size_t)SS * 64 * 64); _Float16* wfh = cvH((size_t)SS * 64 * 64);
  _Float16* atth  = cvH((size_t)BB * SS * 64);
  _Float16* prodh = cvH((size_t)BB * SS * 64);
  float* xbuf  = cvF(MT * 64);
  float* oproj = cvF(MT * 64);
  float* gxb   = cvF(MN * 256);
  float* tsemb = cvF(BB * 64);
  float* nemb  = cvF(BB * 64);
  float* Qb    = cvF((size_t)BB * SS * 64);
  float* Kb    = cvF((size_t)BB * SS * 64);
  float* Vb    = cvF((size_t)BB * SS * 64);
  float* gateb = cvF((size_t)BB * SS * 64);
  float* unstb = cvF((size_t)BB * SS * 64);
  float* fusedb= cvF((size_t)BB * SS * 64);

  // ---- weight / input f16 conversion ----
  cvt(stream, ts_input, tsh, (int)MT, 32, 32);
  cvt(stream, W_inp, wiph, 64, 32, 32);
  cvt(stream, Wqkv, wqkvh, 2 * 192, 64, 64);
  cvt(stream, Wo, woh, 2 * 64, 64, 64);
  cvt(stream, W1, w1h, 2 * DFFN, 64, 64);
  cvt(stream, W2, w2h, 2 * 64, DFFN, DFFN);
  cvt(stream, W_news, wnewsh, 64, 868, 896);   // zero-padded K
  cvt(stream, W_ih, wihh, 256, 64, 64);
  cvt(stream, W_hh, whhh, 256, 64, 64);
  cvt(stream, Wq, wqh, SS * 64, 64, 64);
  cvt(stream, Wk, wkh, SS * 64, 64, 64);
  cvt(stream, Wv, wvh, SS * 64, 64, 64);
  cvt(stream, Wg, wgh, SS * 64, 64, 64);
  cvt(stream, Wu, wuh, SS * 64, 64, 64);
  cvt(stream, Wf, wfh, SS * 64, 64, 64);
  {
    size_t n = MN * 896;
    k_pack_news<<<(int)((n + 255) / 256), 256, 0, stream>>>(news_in, count_in, newsh);
  }

  // ---- transformer trunk ----
  gemm(stream, tsh, wiph, b_inp, xbuf, nullptr, (int)MT, 64, 32, 32, 32, 64, 0, 0, 0, 0, 1, 0);
  {
    size_t n = MT * 64;
    k_pe<<<(int)((n + 255) / 256), 256, 0, stream>>>(xbuf, xh);
  }
  for (int l = 0; l < 2; ++l) {
    gemm(stream, xh, wqkvh + (size_t)l * 192 * 64, bqkv + l * 192, nullptr, qkvh,
         (int)MT, 192, 64, 64, 64, 192, 0, 0, 0, 0, 1, 0);
    k_attn<<<dim3(TT / 16, 4, BB), 32, 0, stream>>>(qkvh, ohb);
    gemm(stream, ohb, woh + (size_t)l * 64 * 64, bo + l * 64, oproj, nullptr,
         (int)MT, 64, 64, 64, 64, 64, 0, 0, 0, 0, 1, 0);
    k_add_ln<<<(int)(MT / 8), 256, 0, stream>>>(xbuf, oproj, ln1_g + l * 64, ln1_b + l * 64,
                                                xbuf, xh, (int)MT);
    k_ffn<<<(int)(MT / 16 / 8), 256, 0, stream>>>(xh, w1h + (size_t)l * DFFN * 64,
                                                  b1 + l * DFFN, w2h + (size_t)l * 64 * DFFN,
                                                  b2 + l * 64, oproj, (int)MT);
    k_add_ln<<<(int)(MT / 8), 256, 0, stream>>>(xbuf, oproj, ln2_g + l * 64, ln2_b + l * 64,
                                                xbuf, xh, (int)MT);
  }
  k_extract<<<(BB * 64 + 255) / 256, 256, 0, stream>>>(xbuf, tsemb, tsembh);

  // ---- news LSTM branch ----
  gemm(stream, newsh, wnewsh, b_news, nullptr, nprojh, (int)MN, 64, 896, 896, 896, 64, 0, 0, 0, 0, 1, 0);
  gemm(stream, nprojh, wihh, b_ih, gxb, nullptr, (int)MN, 256, 64, 64, 64, 256, 0, 0, 0, 0, 1, 0);
  k_lstm<<<BB / 16, 32, 0, stream>>>(gxb, whhh, b_hh, nemb, nembh);

  // ---- fusion: batched per-stock GEMMs (z = S) ----
  gemm(stream, tsembh, wqh, bq, Qb, nullptr, BB, 64, 64, 64, 64, (long long)SS * 64,
       0, 64 * 64, 64, 64, SS, 0);
  gemm(stream, nembh, wkh, bk, Kb, nullptr, BB, 64, 64, 64, 64, (long long)SS * 64,
       0, 64 * 64, 64, 64, SS, 0);
  gemm(stream, nembh, wvh, bv, Vb, nullptr, BB, 64, 64, 64, 64, (long long)SS * 64,
       0, 64 * 64, 64, 64, SS, 0);
  gemm(stream, tsembh, wgh, bg, gateb, nullptr, BB, 64, 64, 64, 64, (long long)SS * 64,
       0, 64 * 64, 64, 64, SS, 2);  // sigmoid
  k_fuse_attn<<<(BB * SS * 4 + 255) / 256, 256, 0, stream>>>(Qb, Kb, Vb, atth);
  gemm(stream, atth, wuh, bu, unstb, nullptr, BB, 64, 64, (long long)SS * 64, 64,
       (long long)SS * 64, 64, 64 * 64, 64, 64, SS, 0);
  k_gate_mul<<<(BB * SS * 64 + 255) / 256, 256, 0, stream>>>(unstb, gateb, prodh);
  gemm(stream, prodh, wfh, bf, fusedb, nullptr, BB, 64, 64, (long long)SS * 64, 64,
       (long long)SS * 64, 64, 64 * 64, 64, 64, SS, 1);  // relu
  k_head<<<(BB * SS + 255) / 256, 256, 0, stream>>>(fusedb, W_head, b_head, out);
}
